// GCN_generator_21921513079200
// MI455X (gfx1250) — compile-verified, matrix operand order checked by reading, not run on hardware
//
#include <hip/hip_runtime.h>
#include <hip/hip_bf16.h>

// ---------------------------------------------------------------------------
// GCN graph generator for MI455X (gfx1250, wave32, WMMA).
//
// One persistent workgroup per batch (8 blocks x 512 threads); the whole
// 127-step sequential chain runs out of LDS. GEMMs use
// v_wmma_f32_16x16x32_f16 with fp32 accumulation; every WMMA operand
// fragment is fetched with two contiguous 16-byte ds_load_b128 from f16
// staging arrays laid out per-operand (row-major for A, transposed for B),
// eliminating the strided scalar-load/wait chains of the previous version.
// Row-normalization is folded into the A operand (H = A*diag(s)*X_unnorm),
// so the transposed X staging is only ever written contiguously.
// ---------------------------------------------------------------------------

typedef __attribute__((ext_vector_type(16))) _Float16 v16h;
typedef __attribute__((ext_vector_type(8)))  _Float16 v8h;
typedef __attribute__((ext_vector_type(4)))  _Float16 v4h;
typedef __attribute__((ext_vector_type(8)))  float    v8f;
typedef __attribute__((ext_vector_type(4)))  float    v4f;

#define GN  128           // N == F == 128
#define GNN (GN * GN)
#define NT  512           // threads per block (16 waves)

// 16x32 f16 fragment, contiguous in K for a fixed row index.
// CDNA5 16-bit operand layout: lanes 0-15 carry K 0..7 / 16..23,
// lanes 16-31 carry K 8..15 / 24..31 (same row set).
// For A operands: row = tile row (row-major source).
// For B operands: row = tile column (transposed source), K contiguous.
__device__ inline v16h load_frag(const _Float16* __restrict__ src, int row, int k0, int lane) {
  const int kb = k0 + ((lane >> 4) << 3);
  const _Float16* p = src + row * GN + kb;
  union { v16h v; v8h h8[2]; } u;
  u.h8[0] = *(const v8h*)(p);        // K kb .. kb+7   -> ds_load_b128
  u.h8[1] = *(const v8h*)(p + 16);   // K kb+16.. +23  -> ds_load_b128
  return u.v;
}

__global__ __launch_bounds__(NT, 1)
void gcn_generate_kernel(const float* __restrict__ x_in,
                         const float* __restrict__ W,
                         const float* __restrict__ bvec,
                         float* __restrict__ out)
{
  extern __shared__ char smem[];
  float*    ADJ = (float*)smem;                 // 64KB carried normalized adjacency (fp32 master)
  float*    X   = ADJ + GNN;                    // 64KB carried features, normalized (fp32 master)
  _Float16* AH  = (_Float16*)(X + GNN);         // 32KB A1: adj * diag(s)   (f16, row-major)
  _Float16* XT  = AH + GNN;                     // 32KB B1: X^T unnormalized (f16, [feat][node])
  _Float16* HH  = XT + GNN;                     // 32KB A2: H = adj@x       (f16, row-major)
  _Float16* WT  = HH + GNN;                     // 32KB B2: W^T             (f16, [out][in])
  float*    BIAS = (float*)(WT + GNN);          // 512B
  float*    PROB = BIAS + GN;                   // 512B
  float*    DV   = PROB + GN;                   // 512B rsqrt(deg)
  float*    XL   = DV + GN;                     // 512B x_last
  float*    SV   = XL + GN;                     // 512B per-row norm scale (folded into AH)

  const int tid  = threadIdx.x;
  const int lane = tid & 31;
  const int wave = tid >> 5;                    // 16 waves
  const int row0 = (wave >> 1) * 16;            // two waves share a 16-row block
  const int ctb  = (wave & 1) * 4;              // ...and split the 8 column tiles
  const int b    = blockIdx.x;
  const float* __restrict__ xb   = x_in + b * GNN;
  float* __restrict__       outb = out  + b * GNN;

  // ---- one-time init ------------------------------------------------------
  for (int e = tid; e < GNN; e += NT) {
    const int r = e >> 7, c = e & (GN - 1);
    X[e]    = xb[e];
    XT[e]   = (_Float16)xb[c * GN + r];         // X^T
    WT[e]   = (_Float16)W[c * GN + r];          // W^T
    ADJ[e]  = (r == c) ? 1.0f : 0.0f;
    outb[e] = (r == c) ? 1.0f : 0.0f;           // adj_output starts as identity
  }
  if (tid < GN) { BIAS[tid] = bvec[tid]; SV[tid] = 1.0f; }
  __syncthreads();

  // i == 0: initial preprocess + GEMM pair (no prob/update/normalize).
  // i >= 1: full generator step.
  for (int i = 0; i < GN; ++i) {
    if (i > 0) {
      // -------- prob = sigmoid((x @ x_i) / 2) ------------------------------
      if (tid < 32) ((v4f*)XL)[tid] = ((const v4f*)(X + i * GN))[tid];
      __syncthreads();
      if (tid < GN) {
        const v4f* xr = (const v4f*)(X + tid * GN);
        const v4f* xl = (const v4f*)XL;
        float s = 0.0f;
#pragma unroll 8
        for (int j = 0; j < GN / 4; ++j) {
          const v4f a = xr[j], v = xl[j];
          s = fmaf(a[0], v[0], fmaf(a[1], v[1], fmaf(a[2], v[2], fmaf(a[3], v[3], s))));
        }
        PROB[tid] = 1.0f / (1.0f + __expf(-0.5f * s));
      }
      __syncthreads();
      // -------- masked row/col write (j < i only, matching set_row_col) ----
      if (tid < i) {
        const float p = PROB[tid];
        ADJ[i * GN + tid]  = p;
        ADJ[tid * GN + i]  = p;
        outb[i * GN + tid] = p;
        outb[tid * GN + i] = p;
      }
      __syncthreads();
    }

    // -------- preprocess: Anorm = d*(A+I)*d (in place) + AH = Anorm*diag(s)
    if (tid < GN) {
      float s = 1.0f;                            // +I diagonal contribution
      const v4f* row = (const v4f*)(ADJ + tid * GN);
#pragma unroll 8
      for (int j = 0; j < GN / 4; ++j) { const v4f a = row[j]; s += a[0] + a[1] + a[2] + a[3]; }
      DV[tid] = rsqrtf(s + 1e-8f);               // entries >= 0 -> isinf guard never fires
    }
    __syncthreads();
    for (int e4 = tid; e4 < GNN / 4; e4 += NT) {
      const int e = e4 * 4;
      const int r = e >> 7, c0 = e & (GN - 1);
      v4f a = *(const v4f*)(ADJ + e);
      const float dr = DV[r];
      const v4f dc = *(const v4f*)(DV + c0);
      const v4f sv = *(const v4f*)(SV + c0);
      v4h ah;
#pragma unroll
      for (int j = 0; j < 4; ++j) {
        const float v = (a[j] + ((r == c0 + j) ? 1.0f : 0.0f)) * dr * dc[j];
        a[j]  = v;
        ah[j] = (_Float16)(v * sv[j]);           // fold row-norm scale of X into A
      }
      *(v4f*)(ADJ + e) = a;
      *(v4h*)(AH + e)  = ah;
    }
    __syncthreads();

    // -------- GEMM 1: H = (Anorm*diag(s)) @ X_unnorm -----------------------
    {
      const v16h a0 = load_frag(AH, row0 + (lane & 15),  0, lane);
      const v16h a1 = load_frag(AH, row0 + (lane & 15), 32, lane);
      const v16h a2 = load_frag(AH, row0 + (lane & 15), 64, lane);
      const v16h a3 = load_frag(AH, row0 + (lane & 15), 96, lane);
#pragma unroll
      for (int t = 0; t < 4; ++t) {
        const int col0 = (ctb + t) * 16;
        const int bn   = col0 + (lane & 15);
        v8f acc = {};
        acc = __builtin_amdgcn_wmma_f32_16x16x32_f16(false, a0, false, load_frag(XT, bn,  0, lane), (short)0, acc, false, false);
        acc = __builtin_amdgcn_wmma_f32_16x16x32_f16(false, a1, false, load_frag(XT, bn, 32, lane), (short)0, acc, false, false);
        acc = __builtin_amdgcn_wmma_f32_16x16x32_f16(false, a2, false, load_frag(XT, bn, 64, lane), (short)0, acc, false, false);
        acc = __builtin_amdgcn_wmma_f32_16x16x32_f16(false, a3, false, load_frag(XT, bn, 96, lane), (short)0, acc, false, false);
        // C/D layout: M = g + 8*(lane>=16), N = lane&15
        const int c  = col0 + (lane & 15);
        const int rb = row0 + ((lane >> 4) << 3);
#pragma unroll
        for (int g = 0; g < 8; ++g) HH[(rb + g) * GN + c] = (_Float16)acc[g];
      }
    }
    __syncthreads();

    // -------- GEMM 2: X = relu(H @ W + b) ----------------------------------
    {
      const v16h a0 = load_frag(HH, row0 + (lane & 15),  0, lane);
      const v16h a1 = load_frag(HH, row0 + (lane & 15), 32, lane);
      const v16h a2 = load_frag(HH, row0 + (lane & 15), 64, lane);
      const v16h a3 = load_frag(HH, row0 + (lane & 15), 96, lane);
#pragma unroll
      for (int t = 0; t < 4; ++t) {
        const int col0 = (ctb + t) * 16;
        const int bn   = col0 + (lane & 15);
        v8f acc = {};
        acc = __builtin_amdgcn_wmma_f32_16x16x32_f16(false, a0, false, load_frag(WT, bn,  0, lane), (short)0, acc, false, false);
        acc = __builtin_amdgcn_wmma_f32_16x16x32_f16(false, a1, false, load_frag(WT, bn, 32, lane), (short)0, acc, false, false);
        acc = __builtin_amdgcn_wmma_f32_16x16x32_f16(false, a2, false, load_frag(WT, bn, 64, lane), (short)0, acc, false, false);
        acc = __builtin_amdgcn_wmma_f32_16x16x32_f16(false, a3, false, load_frag(WT, bn, 96, lane), (short)0, acc, false, false);
        const int c  = col0 + (lane & 15);
        const int rb = row0 + ((lane >> 4) << 3);
        v8h xt;
#pragma unroll
        for (int g = 0; g < 8; ++g) {
          const float v = fmaxf(acc[g] + BIAS[c], 0.0f);
          X[(rb + g) * GN + c] = v;              // fp32 master (normalized below)
          xt[g] = (_Float16)v;                   // unnormalized X^T staging
        }
        // one contiguous 16B store: XT[c][rb..rb+7]
        *(v8h*)(XT + c * GN + rb) = xt;
      }
    }
    __syncthreads();

    // -------- row normalization (steps i >= 1): scale fp32 master only,
    //          record scale in SV -> folded into AH next preprocess ---------
    if (i > 0) {
      if (tid < GN) {
        v4f* xr = (v4f*)(X + tid * GN);
        float s = 0.0f;
#pragma unroll 8
        for (int j = 0; j < GN / 4; ++j) {
          const v4f a = xr[j];
          s = fmaf(a[0], a[0], fmaf(a[1], a[1], fmaf(a[2], a[2], fmaf(a[3], a[3], s))));
        }
        const float inv = 1.0f / (sqrtf(s) + 1e-8f);
#pragma unroll 8
        for (int j = 0; j < GN / 4; ++j) xr[j] *= inv;
        SV[tid] = inv;
      }
      __syncthreads();
    }
  }
}

extern "C" void kernel_launch(void* const* d_in, const int* in_sizes, int n_in,
                              void* d_out, int out_size, void* d_ws, size_t ws_size,
                              hipStream_t stream) {
  (void)in_sizes; (void)n_in; (void)out_size; (void)d_ws; (void)ws_size;
  const float* x = (const float*)d_in[0];   // (8,128,128) f32
  const float* W = (const float*)d_in[1];   // (128,128)   f32
  const float* b = (const float*)d_in[2];   // (128,)      f32
  float* out = (float*)d_out;               // (8,128,128) f32

  // LDS: 2 x 64KB fp32 masters + 4 x 32KB f16 staging + 5 x 512B vectors
  const size_t smem = (size_t)(2 * GNN) * sizeof(float)
                    + (size_t)(4 * GNN) * sizeof(_Float16)
                    + (size_t)(5 * GN) * sizeof(float);
  gcn_generate_kernel<<<dim3(8), dim3(NT), smem, stream>>>(x, W, b, out);
}
